// LSTM_10041633538496
// MI455X (gfx1250) — matrix operand order repaired and checked by reading, not compiled, not run on hardware
//
#include <hip/hip_runtime.h>
#include <hip/hip_bf16.h>

// LSTM (B=512, S=128, IN=1, H=512, OUT=10) for MI455X / gfx1250.
// h,c are [512,512] (faithful to reference). Per step: 4x GEMM 512^3 via
// v_wmma_f32_16x16x32_bf16 with double-buffered LDS tiles (2 x 80KB dynamic
// LDS/block) filled by CDNA5 async copies (global_load_async_to_lds_b128,
// ASYNCcnt), overlapping the global->LDS stream with WMMA compute.
// One wave computes 2 M-tiles x 1 N-tile x 4 gates (8 f32 accumulators),
// then does the wave-local elementwise LSTM update. 128 sequential step
// launches (h ping-pong), then a tiny projection+softmax kernel.

#define Bsz 512
#define Ssz 128
#define Hsz 512
#define OUTsz 10

#define BM 64                                   // block M rows
#define BN 64                                   // block N cols
#define KC 128                                  // K slice staged in LDS
#define ABYTES (BM * KC * 2)                    // 16384
#define SLICE  ((BM * KC + 4 * BN * KC) * 2)    // 81920 bytes per buffer
#define SMEM_BYTES (2 * SLICE)                  // 163840 bytes (double buffer)

typedef __attribute__((ext_vector_type(8)))  float  v8f;
typedef __attribute__((ext_vector_type(8)))  __bf16 v8bf;
typedef __attribute__((ext_vector_type(16))) __bf16 v16bf;

static __device__ __forceinline__ __bf16 f2bf(float f) {
    unsigned int u = __float_as_uint(f);
    unsigned int r = (u + 0x7FFFu + ((u >> 16) & 1u)) >> 16;   // RNE
    unsigned short s = (unsigned short)r;
    __bf16 b;
    __builtin_memcpy(&b, &s, 2);
    return b;
}

static __device__ __forceinline__ float bf2f(__bf16 b) {
    unsigned short s;
    __builtin_memcpy(&s, &b, 2);
    unsigned int u = ((unsigned int)s) << 16;
    return __uint_as_float(u);
}

static __device__ __forceinline__ float fast_sig(float x) {
    return 1.0f / (1.0f + __expf(-x));
}
static __device__ __forceinline__ float fast_tanh(float x) {
    return 2.0f * fast_sig(2.0f * x) - 1.0f;
}

// CDNA5 async copy: 16B per lane, global -> LDS, tracked by ASYNCcnt.
// lds_addr is the absolute LDS byte address (workgroup LDS space).
static __device__ __forceinline__ void async_cp16(unsigned lds_addr, const void* gaddr) {
    asm volatile("global_load_async_to_lds_b128 %0, %1, off"
                 :: "v"(lds_addr), "v"(gaddr)
                 : "memory");
}
static __device__ __forceinline__ void wait_async_le(void) {   // prev chunk done
    asm volatile("s_wait_asynccnt 0x14" ::: "memory");         // <= 20 outstanding
}
static __device__ __forceinline__ void wait_async_all(void) {
    asm volatile("s_wait_asynccnt 0x0" ::: "memory");
}

// ---------------------------------------------------------------------------
// Init: convert recurrent weights fp32->bf16, zero h0 (bf16) and c0 (fp32).
// ---------------------------------------------------------------------------
__global__ void lstm_init(const float* __restrict__ Wgh, const float* __restrict__ Wih,
                          const float* __restrict__ Wfh, const float* __restrict__ Woh,
                          __bf16* __restrict__ wg, __bf16* __restrict__ wi,
                          __bf16* __restrict__ wf, __bf16* __restrict__ wo,
                          __bf16* __restrict__ h0, float* __restrict__ c0) {
    int i = blockIdx.x * blockDim.x + threadIdx.x;
    if (i < Hsz * Hsz) {
        wg[i] = f2bf(Wgh[i]);
        wi[i] = f2bf(Wih[i]);
        wf[i] = f2bf(Wfh[i]);
        wo[i] = f2bf(Woh[i]);
        h0[i] = f2bf(0.0f);
        c0[i] = 0.0f;
    }
}

// ---------------------------------------------------------------------------
// One LSTM time step, double-buffered LDS tiling with async copies.
// Grid: 64 blocks x 256 threads (8 waves). Block tile: [BM=64, BN=64] x 4
// gates. Wave w: M-group (w>>2) covering 2 16-row tiles, N-tile (w&3).
// K sliced into 4 chunks of KC=128; chunk kc+1 streams into the alternate
// LDS buffer while chunk kc is consumed by WMMA.
// ---------------------------------------------------------------------------
__global__ __launch_bounds__(256) void lstm_step(
    const __bf16* __restrict__ hprev, __bf16* __restrict__ hnext,
    float* __restrict__ c,
    const __bf16* __restrict__ wg, const __bf16* __restrict__ wi,
    const __bf16* __restrict__ wf, const __bf16* __restrict__ wo,
    const float* __restrict__ x,
    const float* __restrict__ Wgx, const float* __restrict__ bgx, const float* __restrict__ bgh,
    const float* __restrict__ Wix, const float* __restrict__ bix, const float* __restrict__ bih,
    const float* __restrict__ Wfx, const float* __restrict__ bfx, const float* __restrict__ bfh,
    const float* __restrict__ Wox, const float* __restrict__ box_, const float* __restrict__ boh,
    int t)
{
    extern __shared__ __align__(64) char smem[];
    // Absolute LDS byte address of our dynamic allocation (keeps the asm
    // destination consistent with generic-pointer ds_load reads below).
    const unsigned lds_base =
        (unsigned)(unsigned long long)(__attribute__((address_space(3))) char*)smem;

    const int tid  = threadIdx.x;
    const int lane = tid & 31;
    const int w    = tid >> 5;           // wave 0..7
    const int wmg  = w >> 2;             // M group: rows wmg*32 .. +32
    const int wn   = w & 3;              // N tile
    const int hs   = lane >> 4;          // half-wave select
    const int ln   = lane & 15;
    const int rowBase = (int)(blockIdx.x >> 3) * BM;
    const int colBase = (int)(blockIdx.x & 7) * BN;

    const __bf16* wptr[4] = { wg, wi, wf, wo };

    // Issue the 20 async 16B transfers (4 A + 16 B units) for one K chunk.
    auto issue_chunk = [&](int k0, unsigned bufbase) {
        #pragma unroll
        for (int u = 0; u < 4; ++u) {
            const int unit = tid + u * 256;           // 0..1023
            const int r    = unit >> 4;               // 0..63
            const int kk   = (unit & 15) * 8;         // 0..120
            async_cp16(lds_base + bufbase + (unsigned)((r * KC + kk) * 2),
                       hprev + (rowBase + r) * Hsz + k0 + kk);
        }
        #pragma unroll
        for (int g = 0; g < 4; ++g) {
            const __bf16* wsrc = wptr[g];
            #pragma unroll
            for (int u = 0; u < 4; ++u) {
                const int unit = tid + u * 256;
                const int n    = unit >> 4;
                const int kk   = (unit & 15) * 8;
                async_cp16(lds_base + bufbase +
                               (unsigned)(ABYTES + ((g * BN + n) * KC + kk) * 2),
                           wsrc + (colBase + n) * Hsz + k0 + kk);
            }
        }
    };

    v8f acc[2][4];
    #pragma unroll
    for (int mt = 0; mt < 2; ++mt)
        #pragma unroll
        for (int g = 0; g < 4; ++g)
            acc[mt][g] = (v8f){};

    issue_chunk(0, 0u);                                // prologue: chunk 0 -> buf 0

    for (int kc = 0; kc < Hsz / KC; ++kc) {
        if (kc + 1 < Hsz / KC) {
            issue_chunk((kc + 1) * KC, (unsigned)(((kc + 1) & 1) * SLICE));
            wait_async_le();                           // chunk kc landed (in-order)
        } else {
            wait_async_all();
        }
        __syncthreads();                               // all waves' LDS data visible

        const __bf16* Al = (const __bf16*)(smem + (kc & 1) * SLICE);   // [BM][KC]
        const __bf16* Bl = Al + BM * KC;                               // [4][BN][KC]

        #pragma unroll
        for (int kb = 0; kb < KC / 32; ++kb) {
            const int kk = kb * 32;
            v16bf Amt[2];
            #pragma unroll
            for (int mt = 0; mt < 2; ++mt) {
                const int m = (wmg * 2 + mt) * 16 + ln;
                Amt[mt] = *(const v16bf*)(Al + m * KC + kk + 16 * hs);
            }
            const int n = wn * 16 + ln;
            #pragma unroll
            for (int g = 0; g < 4; ++g) {
                v16bf Bg = *(const v16bf*)(Bl + (g * BN + n) * KC + kk + 16 * hs);
                acc[0][g] = __builtin_amdgcn_wmma_f32_16x16x32_bf16(
                                false, Amt[0], false, Bg, (short)0, acc[0][g], false, false);
                acc[1][g] = __builtin_amdgcn_wmma_f32_16x16x32_bf16(
                                false, Amt[1], false, Bg, (short)0, acc[1][g], false, false);
            }
        }
        __syncthreads();   // buffer kc&1 free for chunk kc+2 before next issue
    }

    // ---- epilogue: rank-1 x contribution + biases, activations, c/h update
    const int j = colBase + wn * 16 + ln;
    const float wxg = Wgx[j], bg_ = bgx[j] + bgh[j];
    const float wxi = Wix[j], bi_ = bix[j] + bih[j];
    const float wxf = Wfx[j], bf_ = bfx[j] + bfh[j];
    const float wxo = Wox[j], bo_ = box_[j] + boh[j];

    #pragma unroll
    for (int mt = 0; mt < 2; ++mt) {
        #pragma unroll
        for (int v = 0; v < 8; ++v) {
            const int   brow = rowBase + (wmg * 2 + mt) * 16 + v + 8 * hs;
            const float xv   = x[brow * Ssz + t];
            const float gg = fast_tanh(acc[mt][0][v] + xv * wxg + bg_);
            const float ii = fast_sig (acc[mt][1][v] + xv * wxi + bi_);
            const float ff = fast_sig (acc[mt][2][v] + xv * wxf + bf_);
            const float oo = fast_sig (acc[mt][3][v] + xv * wxo + bo_);
            const int idx = brow * Hsz + j;
            const float cn = gg * ii + c[idx] * ff;
            c[idx]     = cn;
            hnext[idx] = f2bf(fast_tanh(cn) * oo);
        }
    }
}

// ---------------------------------------------------------------------------
// Final projection + softmax: out[b,:] = softmax(h[b,:] @ Wph.T + bph).
// ---------------------------------------------------------------------------
__global__ void lstm_out(const __bf16* __restrict__ h,
                         const float* __restrict__ Wph, const float* __restrict__ bph,
                         float* __restrict__ out) {
    const int b = blockIdx.x;
    const int l = threadIdx.x;
    float p[OUTsz];
    #pragma unroll
    for (int j = 0; j < OUTsz; ++j) p[j] = 0.0f;
    for (int k = l; k < Hsz; k += 32) {
        const float hv = bf2f(h[b * Hsz + k]);
        #pragma unroll
        for (int j = 0; j < OUTsz; ++j) p[j] += hv * Wph[j * Hsz + k];
    }
    #pragma unroll
    for (int j = 0; j < OUTsz; ++j)
        for (int off = 16; off > 0; off >>= 1)
            p[j] += __shfl_down(p[j], off, 32);
    if (l == 0) {
        float lg[OUTsz], mx = -1e30f;
        #pragma unroll
        for (int j = 0; j < OUTsz; ++j) { lg[j] = p[j] + bph[j]; mx = fmaxf(mx, lg[j]); }
        float s = 0.0f;
        #pragma unroll
        for (int j = 0; j < OUTsz; ++j) { lg[j] = __expf(lg[j] - mx); s += lg[j]; }
        const float inv = 1.0f / s;
        #pragma unroll
        for (int j = 0; j < OUTsz; ++j) out[b * OUTsz + j] = lg[j] * inv;
    }
}

// ---------------------------------------------------------------------------
extern "C" void kernel_launch(void* const* d_in, const int* in_sizes, int n_in,
                              void* d_out, int out_size, void* d_ws, size_t ws_size,
                              hipStream_t stream) {
    const float* x   = (const float*)d_in[0];
    const float* Wgx = (const float*)d_in[1];
    const float* bgx = (const float*)d_in[2];
    const float* Wgh = (const float*)d_in[3];
    const float* bgh = (const float*)d_in[4];
    const float* Wix = (const float*)d_in[5];
    const float* bix = (const float*)d_in[6];
    const float* Wih = (const float*)d_in[7];
    const float* bih = (const float*)d_in[8];
    const float* Wfx = (const float*)d_in[9];
    const float* bfx = (const float*)d_in[10];
    const float* Wfh = (const float*)d_in[11];
    const float* bfh = (const float*)d_in[12];
    const float* Wox = (const float*)d_in[13];
    const float* box_= (const float*)d_in[14];
    const float* Woh = (const float*)d_in[15];
    const float* boh = (const float*)d_in[16];
    const float* Wph = (const float*)d_in[17];
    const float* bph = (const float*)d_in[18];

    char* ws = (char*)d_ws;
    const size_t HE = (size_t)Hsz * Hsz;          // 262144 elements
    __bf16* h0 = (__bf16*)ws;                     // 512 KB
    __bf16* h1 = (__bf16*)(ws + HE * 2);          // 512 KB
    float*  c  = (float*)(ws + HE * 4);           // 1 MB
    __bf16* wg = (__bf16*)(ws + HE * 8);          // 4 x 512 KB bf16 weights
    __bf16* wi = wg + HE;
    __bf16* wf = wi + HE;
    __bf16* wo = wf + HE;

    lstm_init<<<(HE + 255) / 256, 256, 0, stream>>>(Wgh, Wih, Wfh, Woh,
                                                    wg, wi, wf, wo, h0, c);

    for (int t = 0; t < Ssz; ++t) {
        __bf16* hp = (t & 1) ? h1 : h0;
        __bf16* hn = (t & 1) ? h0 : h1;
        lstm_step<<<64, 256, SMEM_BYTES, stream>>>(hp, hn, c, wg, wi, wf, wo, x,
                                                   Wgx, bgx, bgh, Wix, bix, bih,
                                                   Wfx, bfx, bfh, Wox, box_, boh, t);
    }
    // S even -> final h lives in h0
    lstm_out<<<Bsz, 32, 0, stream>>>(h0, Wph, bph, (float*)d_out);
}
